// MultiHeadAttention_52621939311068
// MI455X (gfx1250) — compile-verified, matrix operand order checked by reading, not run on hardware
//
#include <hip/hip_runtime.h>
#include <hip/hip_bf16.h>

// ---------------------------------------------------------------------------
// MHA forward for MI455X (gfx1250, wave32, WMMA).
// B=4, S=2048, E=1024, H=16, D=64.  All matmuls via v_wmma_f32_16x16x32_bf16.
// ---------------------------------------------------------------------------

typedef __attribute__((ext_vector_type(16))) __bf16 v16bf;
typedef __attribute__((ext_vector_type(8)))  __bf16 v8bf;
typedef __attribute__((ext_vector_type(4)))  __bf16 v4bf;
typedef __attribute__((ext_vector_type(8)))  float  v8f;

#define BB 4
#define SS 2048
#define EE 1024
#define HH 16
#define DD 64
#define MM (BB * SS)   // 8192 rows

// float -> bf16 (round to nearest even-ish, bit trick; avoids relying on
// __bf16 conversion support)
__device__ __forceinline__ __bf16 f2bf(float f) {
  unsigned u = __builtin_bit_cast(unsigned, f);
  unsigned r = u + 0x7fffu + ((u >> 16) & 1u);
  unsigned short h = (unsigned short)(r >> 16);
  __bf16 out;
  __builtin_memcpy(&out, &h, 2);
  return out;
}

// Load one 16x32 bf16 WMMA A/B fragment. Per ISA 16-bit A layout:
//   lanes 0-15 : row M = lane,    K = {0..7, 16..23}
//   lanes16-31 : row M = lane-16, K = {8..15, 24..31}
// Caller passes p already offset to (row = lane&15)*stride + (lane>>4)*8.
// Elements 0..7 <- p[0..7], elements 8..15 <- p[16..23]  (two 16B loads).
__device__ __forceinline__ v16bf ldfrag(const __bf16* p) {
  v8bf lo = *(const v8bf*)(p);
  v8bf hi = *(const v8bf*)(p + 16);
  v16bf r;
#pragma unroll
  for (int i = 0; i < 8; ++i) { r[i] = lo[i]; r[i + 8] = hi[i]; }
  return r;
}

__device__ __forceinline__ v8f wmma_bf16(v16bf a, v16bf b, v8f c) {
  return __builtin_amdgcn_wmma_f32_16x16x32_bf16(
      /*neg_a=*/false, a, /*neg_b=*/false, b,
      /*c_mod=*/(short)0, c, /*reuse_a=*/false, /*reuse_b=*/false);
}

// ---------------------------------------------------------------------------
// Kernel 1: f32 -> bf16 cast (vectorized x4)
// ---------------------------------------------------------------------------
__global__ void cast_f32_bf16(const float* __restrict__ src,
                              __bf16* __restrict__ dst, int n4) {
  int i = blockIdx.x * blockDim.x + threadIdx.x;
  if (i >= n4) return;
  float4 f = ((const float4*)src)[i];
  v4bf o;
  o[0] = f2bf(f.x); o[1] = f2bf(f.y); o[2] = f2bf(f.z); o[3] = f2bf(f.w);
  ((v4bf*)dst)[i] = o;
}

// ---------------------------------------------------------------------------
// Kernel 2: GEMM  O[M,N] = X[M,K] @ W[N,K]^T   (torch Linear semantics)
// Because B^T rows == W rows, B fragments read W row-major contiguously.
// Each wave computes a 32x64 tile (2x4 WMMA C tiles).
// storemode: 0 = bf16 row-major, 1 = bf16 transposed-per-head vT[b,h,d,s],
//            2 = f32 row-major (final output).
// ---------------------------------------------------------------------------
__global__ void wmma_gemm(const __bf16* __restrict__ X,
                          const __bf16* __restrict__ W,
                          void* __restrict__ Out, int storemode) {
  const int K = EE, N = EE;
  int wave = (blockIdx.x * blockDim.x + threadIdx.x) >> 5;
  int lane = threadIdx.x & 31;
  int mb = wave >> 4;        // M/32 = 256 tiles
  int nb = wave & 15;        // N/64 = 16  tiles

  int row  = lane & 15;
  int hi   = lane >> 4;
  int koff = hi * 8;

  const __bf16* a0p = X + (size_t)(mb * 32 + row) * K + koff;
  const __bf16* a1p = a0p + (size_t)16 * K;
  const __bf16* bp  = W + (size_t)(nb * 64 + row) * K + koff;

  v8f acc[2][4] = {};

  for (int k0 = 0; k0 < K; k0 += 32) {
    __builtin_prefetch(a0p + k0 + 128, 0, 0);   // global_prefetch_b8
    v16bf a0 = ldfrag(a0p + k0);
    v16bf a1 = ldfrag(a1p + k0);
#pragma unroll
    for (int j = 0; j < 4; ++j) {
      v16bf b = ldfrag(bp + (size_t)j * 16 * K + k0);
      acc[0][j] = wmma_bf16(a0, b, acc[0][j]);
      acc[1][j] = wmma_bf16(a1, b, acc[1][j]);
    }
  }

  // C layout: VGPR i holds row (i + 8*hi), lanes 0-15 / 16-31 give N = lane&15
  int colb = nb * 64 + (lane & 15);
#pragma unroll
  for (int mi = 0; mi < 2; ++mi) {
#pragma unroll
    for (int j = 0; j < 4; ++j) {
#pragma unroll
      for (int i = 0; i < 8; ++i) {
        int r = mb * 32 + mi * 16 + i + hi * 8;
        int c = colb + j * 16;
        float v = acc[mi][j][i];
        if (storemode == 0) {
          ((__bf16*)Out)[(size_t)r * N + c] = f2bf(v);
        } else if (storemode == 1) {
          int b = r >> 11, s = r & 2047;           // r = b*S + s
          int h = c >> 6, d = c & 63;              // c = h*64 + d
          ((__bf16*)Out)[(size_t)((b * HH + h) * DD + d) * SS + s] = f2bf(v);
        } else {
          ((float*)Out)[(size_t)r * N + c] = v;
        }
      }
    }
  }
}

// ---------------------------------------------------------------------------
// Kernel 3: flash attention per (b,h,q-tile of 16 rows), one wave each.
//   scores = Qh Kh^T / 8 ; online softmax ; ctx = P Vh
// qh, kh: bf16 [B*S, E] row-major (d contiguous).  vT: bf16 [B,H,D,S].
// ctx out: bf16 [B*S, E].
// ---------------------------------------------------------------------------
__global__ void flash_attn(const __bf16* __restrict__ qh,
                           const __bf16* __restrict__ kh,
                           const __bf16* __restrict__ vT,
                           __bf16* __restrict__ ctx) {
  __shared__ __bf16 Plds[8][16 * 32];   // per-wave 16x32 prob staging

  int wib  = threadIdx.x >> 5;
  int lane = threadIdx.x & 31;
  int wave = blockIdx.x * 8 + wib;

  int qt = wave & 127;        // 128 q-tiles per (b,h)
  int bh = wave >> 7;
  int b  = bh >> 4, h = bh & 15;

  int row  = lane & 15;
  int hi   = lane >> 4;
  int koff = hi * 8;

  // Q fragments for head-dim chunks d=[0,32) and d=[32,64): reused all loop
  const __bf16* qbase =
      qh + (size_t)(b * SS + qt * 16 + row) * EE + h * DD + koff;
  v16bf aq0 = ldfrag(qbase);
  v16bf aq1 = ldfrag(qbase + 32);

  const __bf16* kbase = kh + (size_t)(b * SS + row) * EE + h * DD + koff;
  const __bf16* vbase =
      vT + (size_t)((b * HH + h) * DD + row) * SS + koff;

  __bf16* pl = &Plds[wib][0];

  float m_i[8], l_i[8];
  v8f acc[4] = {};
#pragma unroll
  for (int i = 0; i < 8; ++i) { m_i[i] = -__builtin_inff(); l_i[i] = 0.f; }

  for (int t0 = 0; t0 < SS; t0 += 32) {
    // ---- scores for keys [t0, t0+32): two 16x16 C tiles --------------------
    v8f c0 = {}, c1 = {};
    const __bf16* kp0 = kbase + (size_t)t0 * EE;
    const __bf16* kp1 = kbase + (size_t)(t0 + 16) * EE;
    c0 = wmma_bf16(aq0, ldfrag(kp0), c0);
    c0 = wmma_bf16(aq1, ldfrag(kp0 + 32), c0);
    c1 = wmma_bf16(aq0, ldfrag(kp1), c1);
    c1 = wmma_bf16(aq1, ldfrag(kp1 + 32), c1);

    // ---- online softmax (rows live across 16-lane halves) ------------------
#pragma unroll
    for (int i = 0; i < 8; ++i) {
      float s0 = c0[i] * 0.125f;           // 1/sqrt(64)
      float s1 = c1[i] * 0.125f;
      float mx = fmaxf(s0, s1);
      mx = fmaxf(mx, __shfl_xor(mx, 1));
      mx = fmaxf(mx, __shfl_xor(mx, 2));
      mx = fmaxf(mx, __shfl_xor(mx, 4));
      mx = fmaxf(mx, __shfl_xor(mx, 8));
      float mnew = fmaxf(m_i[i], mx);
      float p0 = __expf(s0 - mnew);
      float p1 = __expf(s1 - mnew);
      float sum = p0 + p1;
      sum += __shfl_xor(sum, 1);
      sum += __shfl_xor(sum, 2);
      sum += __shfl_xor(sum, 4);
      sum += __shfl_xor(sum, 8);
      float corr = __expf(m_i[i] - mnew);
      l_i[i] = l_i[i] * corr + sum;
      m_i[i] = mnew;
#pragma unroll
      for (int j = 0; j < 4; ++j) acc[j][i] *= corr;
      // stage probs row-major in LDS for C-layout -> A-fragment re-striping
      int prow = i + hi * 8;
      pl[prow * 32 + (lane & 15)]      = f2bf(p0);
      pl[prow * 32 + 16 + (lane & 15)] = f2bf(p1);
    }
    asm volatile("s_wait_dscnt 0" ::: "memory");
    __builtin_amdgcn_wave_barrier();

    // ---- P (16x32) @ V chunk: B^T rows are vT rows (contiguous in t) -------
    v16bf pa = ldfrag(pl + row * 32 + koff);
#pragma unroll
    for (int j = 0; j < 4; ++j) {
      v16bf bv = ldfrag(vbase + (size_t)j * 16 * SS + t0);
      acc[j] = wmma_bf16(pa, bv, acc[j]);
    }
    __builtin_amdgcn_wave_barrier();
  }

  // ---- normalize and store context [B*S, E] bf16 ---------------------------
#pragma unroll
  for (int i = 0; i < 8; ++i) {
    float inv = 1.0f / l_i[i];
    int s = qt * 16 + i + hi * 8;
    size_t base = (size_t)(b * SS + s) * EE + h * DD + (lane & 15);
#pragma unroll
    for (int j = 0; j < 4; ++j)
      ctx[base + j * 16] = f2bf(acc[j][i] * inv);
  }
}

// ---------------------------------------------------------------------------
// Host orchestration
// ---------------------------------------------------------------------------
extern "C" void kernel_launch(void* const* d_in, const int* in_sizes, int n_in,
                              void* d_out, int out_size, void* d_ws,
                              size_t ws_size, hipStream_t stream) {
  (void)in_sizes; (void)n_in; (void)out_size; (void)ws_size;

  const float* q  = (const float*)d_in[0];
  const float* k  = (const float*)d_in[1];
  const float* v  = (const float*)d_in[2];
  const float* Wq = (const float*)d_in[3];
  const float* Wk = (const float*)d_in[4];
  const float* Wv = (const float*)d_in[5];
  const float* Wo = (const float*)d_in[6];

  char* ws = (char*)d_ws;
  size_t off = 0;
  auto carve = [&](size_t bytes) {
    char* p = ws + off;
    off += (bytes + 255) & ~(size_t)255;
    return p;
  };
  const size_t actB = (size_t)MM * EE * 2;   // 16 MB bf16 activations
  const size_t wB   = (size_t)EE * EE * 2;   // 2 MB bf16 weights

  __bf16* qb  = (__bf16*)carve(actB);
  __bf16* kb  = (__bf16*)carve(actB);
  __bf16* vb  = (__bf16*)carve(actB);
  __bf16* Wqb = (__bf16*)carve(wB);
  __bf16* Wkb = (__bf16*)carve(wB);
  __bf16* Wvb = (__bf16*)carve(wB);
  __bf16* Wob = (__bf16*)carve(wB);
  __bf16* qhp = (__bf16*)carve(actB);        // [B*S, E]
  __bf16* khp = (__bf16*)carve(actB);        // [B*S, E]
  __bf16* vTp = (__bf16*)carve(actB);        // [B, H, D, S]
  __bf16* ctx = (__bf16*)carve(actB);        // [B*S, E]

  // 1) downconvert
  {
    int n4a = MM * EE / 4;          // 2M vec4 -> 8192 blocks
    int n4w = EE * EE / 4;          // 256K vec4 -> 1024 blocks
    cast_f32_bf16<<<n4a / 256, 256, 0, stream>>>(q, qb, n4a);
    cast_f32_bf16<<<n4a / 256, 256, 0, stream>>>(k, kb, n4a);
    cast_f32_bf16<<<n4a / 256, 256, 0, stream>>>(v, vb, n4a);
    cast_f32_bf16<<<n4w / 256, 256, 0, stream>>>(Wq, Wqb, n4w);
    cast_f32_bf16<<<n4w / 256, 256, 0, stream>>>(Wk, Wkb, n4w);
    cast_f32_bf16<<<n4w / 256, 256, 0, stream>>>(Wv, Wvb, n4w);
    cast_f32_bf16<<<n4w / 256, 256, 0, stream>>>(Wo, Wob, n4w);
  }

  // 2) projections: (M/32)*(N/64) = 4096 waves -> 512 blocks of 8 waves
  wmma_gemm<<<512, 256, 0, stream>>>(qb, Wqb, qhp, 0);
  wmma_gemm<<<512, 256, 0, stream>>>(kb, Wkb, khp, 0);
  wmma_gemm<<<512, 256, 0, stream>>>(vb, Wvb, vTp, 1);   // V stored transposed

  // 3) attention: B*H*(S/16) = 8192 waves -> 1024 blocks
  flash_attn<<<1024, 256, 0, stream>>>(qhp, khp, vTp, ctx);

  // 4) output projection, f32 straight into d_out
  wmma_gemm<<<512, 256, 0, stream>>>(ctx, Wob, d_out, 2);
}
